// Block_72971494359534
// MI455X (gfx1250) — compile-verified, matrix operand order checked by reading, not run on hardware
//
#include <hip/hip_runtime.h>
#include <hip/hip_bf16.h>
#include <stdint.h>

// ---------------------------------------------------------------------------
// Types for CDNA5 WMMA (wave32): bf16 A/B fragments (16 x bf16 = 8 VGPRs),
// f32 C/D accumulator (8 VGPRs).
// ---------------------------------------------------------------------------
typedef __attribute__((ext_vector_type(16))) __bf16 v16bf;
typedef __attribute__((ext_vector_type(8)))  float  v8f;

union Frag { v16bf f; uint4 u[2]; };

__device__ __forceinline__ unsigned short f2bf(float x) {
  unsigned int u = __float_as_uint(x);
  u += 0x7fffu + ((u >> 16) & 1u);          // round-to-nearest-even
  return (unsigned short)(u >> 16);
}
__device__ __forceinline__ float bf2f(unsigned short h) {
  return __uint_as_float(((unsigned int)h) << 16);
}

// ---------------------------------------------------------------------------
// CDNA5 async global->LDS copy (ASYNCcnt-tracked, no VGPR round trip).
// GV mode: 64-bit global address in a VGPR pair, LDS byte address in a VGPR.
// Async ops complete in order -> s_wait_asynccnt N leaves newest N in flight.
// ---------------------------------------------------------------------------
__device__ __forceinline__ void async_b128(unsigned lds_off, const void* gptr) {
  asm volatile("global_load_async_to_lds_b128 %0, %1, off"
               :: "v"(lds_off), "v"((unsigned long long)(uintptr_t)gptr)
               : "memory");
}
__device__ __forceinline__ void wait_async0() {
  asm volatile("s_wait_asynccnt 0" ::: "memory");
}
__device__ __forceinline__ void wait_async4() {
  asm volatile("s_wait_asynccnt 4" ::: "memory");
}

// ---------------------------------------------------------------------------
// fp32 -> bf16 flat convert (weights / misc)
// ---------------------------------------------------------------------------
__global__ void k_cvt(const float* __restrict__ s, unsigned short* __restrict__ d, int n) {
  int i = blockIdx.x * blockDim.x + threadIdx.x;
  if (i < n) d[i] = f2bf(s[i]);
}

// ---------------------------------------------------------------------------
// RMSNorm row kernel: fp32 in -> bf16 out.  One 256-thread block per row.
// ---------------------------------------------------------------------------
__global__ __launch_bounds__(256) void k_rmsnorm(const float* __restrict__ x,
                                                 const float* __restrict__ w,
                                                 unsigned short* __restrict__ out,
                                                 int C) {
  const int row = blockIdx.x;
  const float* xr = x + (size_t)row * C;
  float s = 0.0f;
  for (int c = threadIdx.x; c < C; c += blockDim.x) { float v = xr[c]; s += v * v; }
  __shared__ float red[256];
  red[threadIdx.x] = s;
  __syncthreads();
  for (int o = 128; o > 0; o >>= 1) {
    if ((int)threadIdx.x < o) red[threadIdx.x] += red[threadIdx.x + o];
    __syncthreads();
  }
  const float scale = rsqrtf(red[0] / (float)C + 1e-6f);
  for (int c = threadIdx.x; c < C; c += blockDim.x)
    out[(size_t)row * C + c] = f2bf(xr[c] * scale * w[c]);
}

// ---------------------------------------------------------------------------
// bf16 WMMA GEMM:  D[M,N] = A[M,K] * W[N,K]^T   (M%128, N%128, K%32 == 0)
// 256 threads = 8 waves; block tile 128x128, k-tile 32.
// Waves arranged 2(m) x 4(n); wave tile 64x32 -> 8 WMMAs per k-step with
// 4 A-frag + 2 B-frag reuse.  Double-buffered LDS; tile k+1 streams in via
// async global->LDS b128 while tile k is computed (ASYNCcnt pipeline).
// EPI: 0 = store bf16; 1 = store f32; 2 = f32 store res+acc;
//      3 = f32 accum += rowscale[m]*acc (rowscale NULL => 1.0);
//      4 = f32 store  = rowscale[m]*acc
// ---------------------------------------------------------------------------
#define BM 128
#define BN 128
#define BK 32
#define LSTR 40   // 32 + 8 pad (ushorts); row stride 80B = multiple of 16B

template <int EPI>
__global__ __launch_bounds__(256) void k_gemm(
    const unsigned short* __restrict__ A, const unsigned short* __restrict__ Bw,
    void* __restrict__ Dst, const float* __restrict__ res,
    const float* __restrict__ rowscale, int M, int N, int K) {
  __shared__ __align__(16) unsigned short As[2][BM * LSTR];
  __shared__ __align__(16) unsigned short Bs[2][BN * LSTR];
  const int tid = threadIdx.x;
  const int wave = tid >> 5, lane = tid & 31;
  const int wr = wave & 1, wc = wave >> 1;     // 2 x 4 wave grid
  const int lh = lane >> 4, ln = lane & 15;
  const int m0 = blockIdx.y * BM, n0 = blockIdx.x * BN;
  const int lr = tid >> 2, lc = (tid & 3) * 8; // 64 rows x 4 chunks per pass

  const unsigned short* apg0 = A  + (size_t)(m0 + lr)      * K + lc;
  const unsigned short* apg1 = A  + (size_t)(m0 + lr + 64) * K + lc;
  const unsigned short* bpg0 = Bw + (size_t)(n0 + lr)      * K + lc;
  const unsigned short* bpg1 = Bw + (size_t)(n0 + lr + 64) * K + lc;
  const unsigned la = (unsigned)(uintptr_t)&As[0][lr * LSTR + lc];
  const unsigned lb = (unsigned)(uintptr_t)&Bs[0][lr * LSTR + lc];
  const unsigned BUFA = BM * LSTR * 2;         // byte offset between buffers
  const unsigned BUFB = BN * LSTR * 2;
  const unsigned HALF = 64 * LSTR * 2;         // rows 64..127

  auto issue = [&](int k0, int buf) {
    const unsigned ao = buf ? BUFA : 0u;
    const unsigned bo = buf ? BUFB : 0u;
    async_b128(la + ao,        apg0 + k0);
    async_b128(la + ao + HALF, apg1 + k0);
    async_b128(lb + bo,        bpg0 + k0);
    async_b128(lb + bo + HALF, bpg1 + k0);
  };

  v8f acc[8];
#pragma unroll
  for (int i = 0; i < 8; ++i) acc[i] = (v8f){};

  issue(0, 0);
  const int nk = K / BK;
  for (int kt = 0; kt < nk; ++kt) {
    const int cur = kt & 1;
    __syncthreads();                 // all waves done reading buffer cur^1
    if (kt + 1 < nk) {
      issue((kt + 1) * BK, cur ^ 1); // stream next tile during compute
      wait_async4();                 // oldest 4 (tile kt) have landed
    } else {
      wait_async0();
    }
    __syncthreads();                 // everyone's tile-kt data visible

    const unsigned short* Ac = As[cur];
    const unsigned short* Bc = Bs[cur];
    Frag a0, a1, a2, a3, b0, b1;
    const int ab = (wr * 64 + ln) * LSTR + 8 * lh;
    a0.u[0] = *(const uint4*)&Ac[ab];
    a0.u[1] = *(const uint4*)&Ac[ab + 16];
    a1.u[0] = *(const uint4*)&Ac[ab + 16 * LSTR];
    a1.u[1] = *(const uint4*)&Ac[ab + 16 * LSTR + 16];
    a2.u[0] = *(const uint4*)&Ac[ab + 32 * LSTR];
    a2.u[1] = *(const uint4*)&Ac[ab + 32 * LSTR + 16];
    a3.u[0] = *(const uint4*)&Ac[ab + 48 * LSTR];
    a3.u[1] = *(const uint4*)&Ac[ab + 48 * LSTR + 16];
    const int bb = (wc * 32 + ln) * LSTR + 8 * lh;
    b0.u[0] = *(const uint4*)&Bc[bb];
    b0.u[1] = *(const uint4*)&Bc[bb + 16];
    b1.u[0] = *(const uint4*)&Bc[bb + 16 * LSTR];
    b1.u[1] = *(const uint4*)&Bc[bb + 16 * LSTR + 16];

    acc[0] = __builtin_amdgcn_wmma_f32_16x16x32_bf16(false, a0.f, false, b0.f, (short)0, acc[0], false, false);
    acc[1] = __builtin_amdgcn_wmma_f32_16x16x32_bf16(false, a0.f, false, b1.f, (short)0, acc[1], false, false);
    acc[2] = __builtin_amdgcn_wmma_f32_16x16x32_bf16(false, a1.f, false, b0.f, (short)0, acc[2], false, false);
    acc[3] = __builtin_amdgcn_wmma_f32_16x16x32_bf16(false, a1.f, false, b1.f, (short)0, acc[3], false, false);
    acc[4] = __builtin_amdgcn_wmma_f32_16x16x32_bf16(false, a2.f, false, b0.f, (short)0, acc[4], false, false);
    acc[5] = __builtin_amdgcn_wmma_f32_16x16x32_bf16(false, a2.f, false, b1.f, (short)0, acc[5], false, false);
    acc[6] = __builtin_amdgcn_wmma_f32_16x16x32_bf16(false, a3.f, false, b0.f, (short)0, acc[6], false, false);
    acc[7] = __builtin_amdgcn_wmma_f32_16x16x32_bf16(false, a3.f, false, b1.f, (short)0, acc[7], false, false);
  }

#pragma unroll
  for (int i = 0; i < 4; ++i) {
#pragma unroll
    for (int j = 0; j < 2; ++j) {
      const v8f cc = acc[i * 2 + j];
#pragma unroll
      for (int r = 0; r < 8; ++r) {
        const int m = m0 + wr * 64 + i * 16 + lh * 8 + r;
        const size_t idx = (size_t)m * N + n0 + wc * 32 + j * 16 + ln;
        const float vv = cc[r];
        if (EPI == 0) {
          ((unsigned short*)Dst)[idx] = f2bf(vv);
        } else if (EPI == 1) {
          ((float*)Dst)[idx] = vv;
        } else if (EPI == 2) {
          ((float*)Dst)[idx] = res[idx] + vv;
        } else if (EPI == 3) {
          const float s = rowscale ? rowscale[m] : 1.0f;
          ((float*)Dst)[idx] += s * vv;
        } else {
          ((float*)Dst)[idx] = rowscale[m] * vv;
        }
      }
    }
  }
}

// ---------------------------------------------------------------------------
// RoPE in place on bf16 (tok, nheads*64) buffer.  T=1024, HEAD_DIM=64.
// ---------------------------------------------------------------------------
__global__ void k_rope(unsigned short* __restrict__ t, const float* __restrict__ fc,
                       int nheads, int n) {
  int i = blockIdx.x * blockDim.x + threadIdx.x;
  if (i >= n) return;
  const int d2 = i & 31;
  const int hh = (i >> 5) % nheads;
  const int tok = i / (32 * nheads);
  const int tt = tok & 1023;
  const float cs = fc[(tt * 32 + d2) * 2];
  const float sn = fc[(tt * 32 + d2) * 2 + 1];
  unsigned short* p = t + (size_t)tok * (nheads * 64) + hh * 64 + 2 * d2;
  const float re = bf2f(p[0]), im = bf2f(p[1]);
  p[0] = f2bf(re * cs - im * sn);
  p[1] = f2bf(im * cs + re * sn);
}

// ---------------------------------------------------------------------------
// Flash attention (causal, GQA 16/4, HEAD_DIM=64, T=1024).
// 128 threads = 4 waves; block = (b, h, 64-row q tile); wave = 16 q rows.
// Per 32-key block per wave: 4 WMMA (QK^T) + 4 WMMA (PV).
// K tile staged with async global->LDS; V transposed with scalar stores.
// ---------------------------------------------------------------------------
__global__ __launch_bounds__(128) void k_attn(
    const unsigned short* __restrict__ q, const unsigned short* __restrict__ k,
    const unsigned short* __restrict__ v, unsigned short* __restrict__ o) {
  __shared__ __align__(16) unsigned short Ks[32 * 72];      // [key][d], stride 72
  __shared__ __align__(16) unsigned short Vt[64 * 40];      // [d][key], stride 40
  __shared__ __align__(16) unsigned short Ps[4 * 16 * 40];  // per-wave P transpose

  const int bid = blockIdx.x;
  const int qb = bid & 15, h = (bid >> 4) & 15, b = bid >> 8;
  const int kvh = h >> 2;
  const int tid = threadIdx.x, wave = tid >> 5, lane = tid & 31;
  const int lh = lane >> 4, ln = lane & 15;

  Frag qa0, qa1;  // Q fragments resident in registers for whole kernel
  {
    const size_t tok = (size_t)b * 1024 + qb * 64 + wave * 16 + ln;
    const unsigned short* qp = q + tok * 1024 + h * 64 + 8 * lh;
    qa0.u[0] = *(const uint4*)(qp);
    qa0.u[1] = *(const uint4*)(qp + 16);
    qa1.u[0] = *(const uint4*)(qp + 32);
    qa1.u[1] = *(const uint4*)(qp + 48);
  }

  v8f oa0 = {}, oa1 = {}, oa2 = {}, oa3 = {};
  float mi[8], li[8];
#pragma unroll
  for (int r = 0; r < 8; ++r) { mi[r] = -1e30f; li[r] = 0.0f; }

  unsigned short* pw = &Ps[wave * 16 * 40];
  const int kr_ld = tid >> 2, cc_ld = (tid & 3) * 16;
  const unsigned lds_k = (unsigned)(uintptr_t)&Ks[kr_ld * 72 + cc_ld];
  const int kbmax = 2 * qb + 1;
  for (int kb = 0; kb <= kbmax; ++kb) {
    __syncthreads();  // protect Ks/Vt against previous iteration readers
    {  // K tile 32x64 -> LDS (row-major, stride 72) via async copy
      const size_t tok = (size_t)b * 1024 + kb * 32 + kr_ld;
      const unsigned short* kp = k + tok * 256 + kvh * 64 + cc_ld;
      async_b128(lds_k, kp);
      async_b128(lds_k + 16, kp + 8);
    }
#pragma unroll
    for (int e = 0; e < 16; ++e) {  // V tile transposed: Vt[d][key]
      const int idx = tid * 16 + e;
      const int key = idx >> 6, d = idx & 63;
      const size_t tok = (size_t)b * 1024 + kb * 32 + key;
      Vt[d * 40 + key] = v[tok * 256 + kvh * 64 + d];
    }
    wait_async0();
    __syncthreads();

    // ---- S = Q K^T  (two 16x16 tiles over 32 keys, contract d=64) ----
    Frag kf;
    v8f s0 = {}, s1 = {};
    {
      const int kr = ln * 72 + 8 * lh;
      kf.u[0] = *(const uint4*)&Ks[kr];
      kf.u[1] = *(const uint4*)&Ks[kr + 16];
      s0 = __builtin_amdgcn_wmma_f32_16x16x32_bf16(false, qa0.f, false, kf.f, (short)0, s0, false, false);
      kf.u[0] = *(const uint4*)&Ks[kr + 32];
      kf.u[1] = *(const uint4*)&Ks[kr + 48];
      s0 = __builtin_amdgcn_wmma_f32_16x16x32_bf16(false, qa1.f, false, kf.f, (short)0, s0, false, false);
      const int kr1 = (16 + ln) * 72 + 8 * lh;
      kf.u[0] = *(const uint4*)&Ks[kr1];
      kf.u[1] = *(const uint4*)&Ks[kr1 + 16];
      s1 = __builtin_amdgcn_wmma_f32_16x16x32_bf16(false, qa0.f, false, kf.f, (short)0, s1, false, false);
      kf.u[0] = *(const uint4*)&Ks[kr1 + 32];
      kf.u[1] = *(const uint4*)&Ks[kr1 + 48];
      s1 = __builtin_amdgcn_wmma_f32_16x16x32_bf16(false, qa1.f, false, kf.f, (short)0, s1, false, false);
    }

    // ---- online softmax (per-row reductions across the 16-lane groups) ----
#pragma unroll
    for (int r = 0; r < 8; ++r) {
      const int qg = qb * 64 + wave * 16 + lh * 8 + r;
      const int kg0 = kb * 32 + ln, kg1 = kg0 + 16;
      float x0 = (kg0 <= qg) ? s0[r] * 0.125f : -1e30f;
      float x1 = (kg1 <= qg) ? s1[r] * 0.125f : -1e30f;
      float rm = fmaxf(x0, x1);
      rm = fmaxf(rm, __shfl_xor(rm, 1));
      rm = fmaxf(rm, __shfl_xor(rm, 2));
      rm = fmaxf(rm, __shfl_xor(rm, 4));
      rm = fmaxf(rm, __shfl_xor(rm, 8));
      const float mnew  = fmaxf(mi[r], rm);
      const float alpha = __expf(mi[r] - mnew);
      const float e0 = __expf(x0 - mnew);
      const float e1 = __expf(x1 - mnew);
      float rs = e0 + e1;
      rs += __shfl_xor(rs, 1);
      rs += __shfl_xor(rs, 2);
      rs += __shfl_xor(rs, 4);
      rs += __shfl_xor(rs, 8);
      li[r] = li[r] * alpha + rs;
      mi[r] = mnew;
      oa0[r] *= alpha; oa1[r] *= alpha; oa2[r] *= alpha; oa3[r] *= alpha;
      // C-fragment -> A-fragment layout transpose through per-wave LDS
      pw[(lh * 8 + r) * 40 + ln]      = f2bf(e0);
      pw[(lh * 8 + r) * 40 + 16 + ln] = f2bf(e1);
    }

    // ---- O += P V  (P is a single 16x32 bf16 A-fragment) ----
    Frag pf, vf;
    pf.u[0] = *(const uint4*)&pw[ln * 40 + 8 * lh];
    pf.u[1] = *(const uint4*)&pw[ln * 40 + 16 + 8 * lh];
    vf.u[0] = *(const uint4*)&Vt[ln * 40 + 8 * lh];
    vf.u[1] = *(const uint4*)&Vt[ln * 40 + 16 + 8 * lh];
    oa0 = __builtin_amdgcn_wmma_f32_16x16x32_bf16(false, pf.f, false, vf.f, (short)0, oa0, false, false);
    vf.u[0] = *(const uint4*)&Vt[(16 + ln) * 40 + 8 * lh];
    vf.u[1] = *(const uint4*)&Vt[(16 + ln) * 40 + 16 + 8 * lh];
    oa1 = __builtin_amdgcn_wmma_f32_16x16x32_bf16(false, pf.f, false, vf.f, (short)0, oa1, false, false);
    vf.u[0] = *(const uint4*)&Vt[(32 + ln) * 40 + 8 * lh];
    vf.u[1] = *(const uint4*)&Vt[(32 + ln) * 40 + 16 + 8 * lh];
    oa2 = __builtin_amdgcn_wmma_f32_16x16x32_bf16(false, pf.f, false, vf.f, (short)0, oa2, false, false);
    vf.u[0] = *(const uint4*)&Vt[(48 + ln) * 40 + 8 * lh];
    vf.u[1] = *(const uint4*)&Vt[(48 + ln) * 40 + 16 + 8 * lh];
    oa3 = __builtin_amdgcn_wmma_f32_16x16x32_bf16(false, pf.f, false, vf.f, (short)0, oa3, false, false);
  }

#pragma unroll
  for (int r = 0; r < 8; ++r) {
    const size_t tok = (size_t)b * 1024 + qb * 64 + wave * 16 + lh * 8 + r;
    unsigned short* op = o + tok * 1024 + h * 64 + ln;
    const float inv = 1.0f / li[r];
    op[0]  = f2bf(oa0[r] * inv);
    op[16] = f2bf(oa1[r] * inv);
    op[32] = f2bf(oa2[r] * inv);
    op[48] = f2bf(oa3[r] * inv);
  }
}

// ---------------------------------------------------------------------------
// Gate logits (N=8 too small for WMMA): scores[n][e] = xn2[n] . gate_w[e]
// ---------------------------------------------------------------------------
__global__ void k_gate(const unsigned short* __restrict__ xn2,
                       const float* __restrict__ gw, float* __restrict__ scores) {
  int i = blockIdx.x * blockDim.x + threadIdx.x;
  if (i >= 4096 * 8) return;
  const int e = i & 7, n = i >> 3;
  const unsigned short* xr = xn2 + (size_t)n * 1024;
  const float* wr = gw + (size_t)e * 1024;
  float s = 0.0f;
  for (int c = 0; c < 1024; ++c) s += bf2f(xr[c]) * wr[c];
  scores[i] = s;
}

// softmax over 8 + top-2 -> dense combine weights, stored transposed (E, N)
__global__ void k_topk(const float* __restrict__ scores, float* __restrict__ wfull_t) {
  int n = blockIdx.x * blockDim.x + threadIdx.x;
  if (n >= 4096) return;
  float sc[8]; float mx = -3e38f;
#pragma unroll
  for (int e = 0; e < 8; ++e) { sc[e] = scores[n * 8 + e]; mx = fmaxf(mx, sc[e]); }
  float sum = 0.0f;
#pragma unroll
  for (int e = 0; e < 8; ++e) { sc[e] = __expf(sc[e] - mx); sum += sc[e]; }
  const float inv = 1.0f / sum;
#pragma unroll
  for (int e = 0; e < 8; ++e) sc[e] *= inv;
  int i1 = 0;
#pragma unroll
  for (int e = 1; e < 8; ++e) if (sc[e] > sc[i1]) i1 = e;
  int i2 = (i1 == 0) ? 1 : 0;
#pragma unroll
  for (int e = 0; e < 8; ++e) if (e != i1 && sc[e] > sc[i2]) i2 = e;
#pragma unroll
  for (int e = 0; e < 8; ++e)
    wfull_t[(size_t)e * 4096 + n] = (e == i1 || e == i2) ? sc[e] : 0.0f;
}

// h = silu(g1) * g2  ->  bf16
__global__ void k_silumul(const float* __restrict__ g1, const float* __restrict__ g2,
                          unsigned short* __restrict__ h, int n) {
  int i = blockIdx.x * blockDim.x + threadIdx.x;
  if (i >= n) return;
  const float a = g1[i];
  h[i] = f2bf((a / (1.0f + __expf(-a))) * g2[i]);
}

__global__ void k_add(const float* __restrict__ a, const float* __restrict__ b,
                      float* __restrict__ out, int n) {
  int i = blockIdx.x * blockDim.x + threadIdx.x;
  if (i < n) out[i] = a[i] + b[i];
}

// ---------------------------------------------------------------------------
// Host orchestration
// ---------------------------------------------------------------------------
extern "C" void kernel_launch(void* const* d_in, const int* in_sizes, int n_in,
                              void* d_out, int out_size, void* d_ws, size_t ws_size,
                              hipStream_t stream) {
  (void)in_sizes; (void)n_in; (void)out_size; (void)ws_size;
  const int NTOK = 4096, C = 1024, IE = 1024, IS = 2048;

  const float* x   = (const float*)d_in[0];
  const float* fc  = (const float*)d_in[1];
  const float* n1w = (const float*)d_in[4];
  const float* wq  = (const float*)d_in[5];
  const float* wk  = (const float*)d_in[6];
  const float* wv  = (const float*)d_in[7];
  const float* wo  = (const float*)d_in[8];
  const float* n2w = (const float*)d_in[9];
  const float* gw  = (const float*)d_in[10];
  const float* ew1 = (const float*)d_in[11];
  const float* ew2 = (const float*)d_in[12];
  const float* ewp = (const float*)d_in[13];
  const float* sw1 = (const float*)d_in[14];
  const float* sw2 = (const float*)d_in[15];
  const float* swp = (const float*)d_in[16];

  float* out    = (float*)d_out;                 // (4096,1024)
  float* scores = out + (size_t)NTOK * C;        // (4096,8)

  char* wsb = (char*)d_ws;
  size_t off = 0;
  auto alloc = [&](size_t bytes) -> char* {
    char* p = wsb + off;
    off = (off + bytes + 255) & ~(size_t)255;
    return p;
  };
  unsigned short* wq_b  = (unsigned short*)alloc((size_t)C * C * 2);
  unsigned short* wk_b  = (unsigned short*)alloc((size_t)256 * C * 2);
  unsigned short* wv_b  = (unsigned short*)alloc((size_t)256 * C * 2);
  unsigned short* wo_b  = (unsigned short*)alloc((size_t)C * C * 2);
  unsigned short* ew1_b = (unsigned short*)alloc((size_t)8 * IE * C * 2);
  unsigned short* ew2_b = (unsigned short*)alloc((size_t)8 * IE * C * 2);
  unsigned short* ewp_b = (unsigned short*)alloc((size_t)8 * C * IE * 2);
  unsigned short* sw1_b = (unsigned short*)alloc((size_t)IS * C * 2);
  unsigned short* sw2_b = (unsigned short*)alloc((size_t)IS * C * 2);
  unsigned short* swp_b = (unsigned short*)alloc((size_t)C * IS * 2);
  unsigned short* xn    = (unsigned short*)alloc((size_t)NTOK * C * 2);
  unsigned short* qb    = (unsigned short*)alloc((size_t)NTOK * C * 2);
  unsigned short* kb    = (unsigned short*)alloc((size_t)NTOK * 256 * 2);
  unsigned short* vb    = (unsigned short*)alloc((size_t)NTOK * 256 * 2);
  unsigned short* ob    = (unsigned short*)alloc((size_t)NTOK * C * 2);
  float*          xres  = (float*)alloc((size_t)NTOK * C * 4);
  unsigned short* xn2   = (unsigned short*)alloc((size_t)NTOK * C * 2);
  float*          g1    = (float*)alloc((size_t)NTOK * IS * 4);
  float*          g2    = (float*)alloc((size_t)NTOK * IS * 4);
  unsigned short* hb    = (unsigned short*)alloc((size_t)NTOK * IS * 2);
  float*          yacc  = (float*)alloc((size_t)NTOK * C * 4);
  float*          wft   = (float*)alloc((size_t)8 * NTOK * 4);

  auto conv = [&](const float* s, unsigned short* d, int n) {
    k_cvt<<<dim3((n + 255) / 256), dim3(256), 0, stream>>>(s, d, n);
  };
  auto gemm = [&](const unsigned short* A, const unsigned short* Bw, void* D,
                  const float* res, const float* rs, int M, int N, int K, int epi) {
    dim3 g(N / 128, M / 128);
    switch (epi) {
      case 0: k_gemm<0><<<g, 256, 0, stream>>>(A, Bw, D, res, rs, M, N, K); break;
      case 1: k_gemm<1><<<g, 256, 0, stream>>>(A, Bw, D, res, rs, M, N, K); break;
      case 2: k_gemm<2><<<g, 256, 0, stream>>>(A, Bw, D, res, rs, M, N, K); break;
      case 3: k_gemm<3><<<g, 256, 0, stream>>>(A, Bw, D, res, rs, M, N, K); break;
      default: k_gemm<4><<<g, 256, 0, stream>>>(A, Bw, D, res, rs, M, N, K); break;
    }
  };

  // weight conversion
  conv(wq, wq_b, C * C);
  conv(wk, wk_b, 256 * C);
  conv(wv, wv_b, 256 * C);
  conv(wo, wo_b, C * C);
  conv(ew1, ew1_b, 8 * IE * C);
  conv(ew2, ew2_b, 8 * IE * C);
  conv(ewp, ewp_b, 8 * C * IE);
  conv(sw1, sw1_b, IS * C);
  conv(sw2, sw2_b, IS * C);
  conv(swp, swp_b, C * IS);

  // ---- attention ----
  k_rmsnorm<<<dim3(NTOK), dim3(256), 0, stream>>>(x, n1w, xn, C);
  gemm(xn, wq_b, qb, nullptr, nullptr, NTOK, C, C, 0);
  gemm(xn, wk_b, kb, nullptr, nullptr, NTOK, 256, C, 0);
  gemm(xn, wv_b, vb, nullptr, nullptr, NTOK, 256, C, 0);
  {
    int nq = NTOK * 16 * 32, nk = NTOK * 4 * 32;
    k_rope<<<dim3((nq + 255) / 256), dim3(256), 0, stream>>>(qb, fc, 16, nq);
    k_rope<<<dim3((nk + 255) / 256), dim3(256), 0, stream>>>(kb, fc, 4, nk);
  }
  k_attn<<<dim3(1024), dim3(128), 0, stream>>>(qb, kb, vb, ob);
  gemm(ob, wo_b, xres, x, nullptr, NTOK, C, C, 2);   // xres = x + o @ wo^T

  // ---- MoE ----
  k_rmsnorm<<<dim3(NTOK), dim3(256), 0, stream>>>(xres, n2w, xn2, C);
  k_gate<<<dim3((NTOK * 8 + 255) / 256), dim3(256), 0, stream>>>(xn2, gw, scores);
  k_topk<<<dim3((NTOK + 255) / 256), dim3(256), 0, stream>>>(scores, wft);

  for (int e = 0; e < 8; ++e) {
    const unsigned short* w1e = ew1_b + (size_t)e * IE * C;
    const unsigned short* w2e = ew2_b + (size_t)e * IE * C;
    const unsigned short* wpe = ewp_b + (size_t)e * C * IE;
    gemm(xn2, w1e, g1, nullptr, nullptr, NTOK, IE, C, 1);
    gemm(xn2, w2e, g2, nullptr, nullptr, NTOK, IE, C, 1);
    int nh = NTOK * IE;
    k_silumul<<<dim3((nh + 255) / 256), dim3(256), 0, stream>>>(g1, g2, hb, nh);
    gemm(hb, wpe, yacc, nullptr, wft + (size_t)e * NTOK, NTOK, C, IE, e == 0 ? 4 : 3);
  }
  // shared expert
  gemm(xn2, sw1_b, g1, nullptr, nullptr, NTOK, IS, C, 1);
  gemm(xn2, sw2_b, g2, nullptr, nullptr, NTOK, IS, C, 1);
  {
    int nh = NTOK * IS;
    k_silumul<<<dim3((nh + 255) / 256), dim3(256), 0, stream>>>(g1, g2, hb, nh);
  }
  gemm(hb, swp_b, yacc, nullptr, nullptr, NTOK, C, IS, 3);

  // out = xres + (y + z)
  {
    int n = NTOK * C;
    k_add<<<dim3((n + 255) / 256), dim3(256), 0, stream>>>(xres, yacc, out, n);
  }
}
